// SelfAttention_21809843929397
// MI455X (gfx1250) — matrix operand order repaired and checked by reading, not compiled, not run on hardware
//
#include <hip/hip_runtime.h>

// ---------------------------------------------------------------------------
// Self-attention (non-local block) for MI455X / gfx1250, wave32 + WMMA f16.
// B=16, C=256, H=W=64 -> N=4096 queries, M=1024 pooled keys.
// All GEMMs via v_wmma_f32_16x16x32_f16; softmax kept fully in LDS.
// Round 2: software-pipelined GEMM inner loops (prefetch into distinct regs
// so WMMAs don't serialize behind s_wait_loadcnt 0 on a shared buffer).
// ---------------------------------------------------------------------------

typedef __attribute__((ext_vector_type(16))) _Float16 v16h;
typedef __attribute__((ext_vector_type(8)))  _Float16 v8h;
typedef __attribute__((ext_vector_type(8)))  float    v8f;

union A16 { v16h v; v8h h[2]; };

#define NB  16
#define NC  256
#define NC8 32
#define NC2 128
#define NN  4096
#define NM  1024
#define NO  192   // 32 (theta) + 32 (phi) + 128 (g)

static __device__ __forceinline__ v8f wmma_f16(v16h a, v16h b, v8f c) {
  // D = A(16x32 f16) * B(32x16 f16) + C(16x16 f32)
  return __builtin_amdgcn_wmma_f32_16x16x32_f16(false, a, false, b, (short)0, c,
                                                false, false);
}

// Workspace layout (bytes), 256B-aligned regions, total ~77.2 MB
constexpr size_t OFF_WCAT  = 0;                        // 192*256 f16   =   98304
constexpr size_t OFF_WOH   = 131072;                   // 256*128 f16   =   65536
constexpr size_t OFF_BCAT  = 196608;                   // 192 f32       =     768
constexpr size_t OFF_XT    = 262144;                   // B*N*C f16     = 33554432
constexpr size_t OFF_THETA = OFF_XT    + 33554432ull;  // B*N*32 f16    =  4194304
constexpr size_t OFF_PROJ  = OFF_THETA + 4194304ull;   // B*N*160 f16   = 20971520
constexpr size_t OFF_PHIT  = OFF_PROJ  + 20971520ull;  // B*M*32 f16    =  1048576
constexpr size_t OFF_G     = OFF_PHIT  + 1048576ull;   // B*128*M f16   =  4194304
constexpr size_t OFF_OPRE  = OFF_G     + 4194304ull;   // B*N*128 f16   = 16777216

// ---------------------------------------------------------------------------
// K0a: weights -> f16.  w_cat[o][c] concatenates theta/phi/g rows (natural
// [out][in] layout == WMMA B-tile layout: lane j = col o, contiguous K=c).
// ---------------------------------------------------------------------------
__global__ void k_prep_weights(const float* __restrict__ wt, const float* __restrict__ wp,
                               const float* __restrict__ wg, const float* __restrict__ bt,
                               const float* __restrict__ bp, const float* __restrict__ bg,
                               const float* __restrict__ wo,
                               _Float16* __restrict__ w_cat, _Float16* __restrict__ wo_h,
                               float* __restrict__ b_cat) {
  int i = blockIdx.x * 256 + threadIdx.x;
  if (i < NO * NC) {
    int o = i / NC, c = i % NC;
    float v = (o < 32) ? wt[o * NC + c]
            : (o < 64) ? wp[(o - 32) * NC + c]
                       : wg[(o - 64) * NC + c];
    w_cat[i] = (_Float16)v;
  }
  if (i < NC * NC2) wo_h[i] = (_Float16)wo[i];
  if (i < NO) b_cat[i] = (i < 32) ? bt[i] : (i < 64) ? bp[i - 32] : bg[i - 64];
}

// ---------------------------------------------------------------------------
// K0b: x [B][C][N] f32 -> xT [B][N][C] f16 (LDS tile transpose, coalesced).
// ---------------------------------------------------------------------------
__global__ void k_transpose_x(const float* __restrict__ x, _Float16* __restrict__ xT) {
  __shared__ float tile[32][33];
  int b  = blockIdx.z;
  int c0 = blockIdx.y * 32, n0 = blockIdx.x * 32;
  int tx = threadIdx.x, ty = threadIdx.y;   // 32 x 8
  #pragma unroll
  for (int j = 0; j < 4; ++j)
    tile[ty + 8 * j][tx] = x[((size_t)b * NC + c0 + ty + 8 * j) * NN + n0 + tx];
  __syncthreads();
  #pragma unroll
  for (int j = 0; j < 4; ++j)
    xT[((size_t)b * NN + n0 + ty + 8 * j) * NC + c0 + tx] = (_Float16)tile[tx][ty + 8 * j];
}

// ---------------------------------------------------------------------------
// K1: fused projection GEMM. outT (N x 192) = xT (N x 256) @ w_cat^T, K=256.
// Block = 128 threads (4 waves); wave handles 16 rows n, all 12 o-tiles.
// Software pipelined: A for next k-step and B for next o-tile prefetched
// into distinct registers before each WMMA (indices wrap; extra loads cheap).
// theta cols 0..31 -> thetaT [B][N][32]; phi/g cols -> projT [B][N][160].
// ---------------------------------------------------------------------------
__global__ void __launch_bounds__(128)
k_proj(const _Float16* __restrict__ xT, const _Float16* __restrict__ w_cat,
       const float* __restrict__ b_cat, _Float16* __restrict__ thetaT,
       _Float16* __restrict__ projT) {
  int tid = threadIdx.x, wave = tid >> 5, lane = tid & 31;
  int b = blockIdx.y;
  int n_base = blockIdx.x * 64 + wave * 16;
  int colL = lane & 15, hi = lane >> 4, rowHi = hi * 8;

  v8f acc[12];
  #pragma unroll
  for (int i = 0; i < 12; ++i) acc[i] = (v8f){0.f,0.f,0.f,0.f,0.f,0.f,0.f,0.f};

  const _Float16* arow  = xT + ((size_t)b * NN + n_base + colL) * NC + hi * 16;
  const _Float16* wbase = w_cat + (size_t)colL * NC + hi * 16;

  auto loadB = [&](int ks, int ot) -> v16h {
    return *(const v16h*)(wbase + (size_t)(ot * 16) * NC + ks * 32);
  };
  auto loadA = [&](int ks, A16& a) {
    a.h[0] = *(const v8h*)(arow + ks * 32);
    a.h[1] = *(const v8h*)(arow + ks * 32 + 8);
  };

  A16 a_cur, a_nxt;
  loadA(0, a_cur);
  v16h b_cur = loadB(0, 0);
  for (int ks = 0; ks < 8; ++ks) {
    loadA((ks + 1) & 7, a_nxt);
    #pragma unroll
    for (int ot = 0; ot < 12; ++ot) {
      v16h b_nxt = (ot < 11) ? loadB(ks, ot + 1) : loadB((ks + 1) & 7, 0);
      acc[ot] = wmma_f16(a_cur.v, b_cur, acc[ot]);
      b_cur = b_nxt;
    }
    a_cur = a_nxt;
  }

  for (int ot = 0; ot < 12; ++ot) {
    int o = ot * 16 + colL;
    float bias = b_cat[o];
    #pragma unroll
    for (int v = 0; v < 8; ++v) {
      float val = acc[ot][v] + bias;
      int n = n_base + rowHi + v;
      if (o < 32) thetaT[((size_t)b * NN + n) * 32 + o] = (_Float16)val;
      else        projT [((size_t)b * NN + n) * 160 + (o - 32)] = (_Float16)val;
    }
  }
}

// ---------------------------------------------------------------------------
// K2a: 2x2 maxpool of phi -> phiT [B][M][32] (B-tile layout for score GEMM).
// ---------------------------------------------------------------------------
__global__ void k_pool_phi(const _Float16* __restrict__ projT, _Float16* __restrict__ phiT) {
  int b = blockIdx.y;
  int o = threadIdx.x;                           // 0..31
  int m = blockIdx.x * 8 + threadIdx.y;          // 0..1023
  int hm = m >> 5, wm = m & 31;
  size_t base = ((size_t)b * NN + (size_t)(2 * hm) * 64 + 2 * wm) * 160 + o;
  float v0 = (float)projT[base];
  float v1 = (float)projT[base + 160];
  float v2 = (float)projT[base + (size_t)64 * 160];
  float v3 = (float)projT[base + (size_t)65 * 160];
  phiT[((size_t)b * NM + m) * 32 + o] = (_Float16)fmaxf(fmaxf(v0, v1), fmaxf(v2, v3));
}

// ---------------------------------------------------------------------------
// K2b: 2x2 maxpool of g -> gmat [B][128][M] (B-tile layout for beta@g GEMM).
// ---------------------------------------------------------------------------
__global__ void k_pool_g(const _Float16* __restrict__ projT, _Float16* __restrict__ gmat) {
  int b = blockIdx.z;
  int m = blockIdx.x * 32 + threadIdx.x;
  int c = blockIdx.y * 8 + threadIdx.y;
  int hm = m >> 5, wm = m & 31;
  size_t base = ((size_t)b * NN + (size_t)(2 * hm) * 64 + 2 * wm) * 160 + 32 + c;
  float v0 = (float)projT[base];
  float v1 = (float)projT[base + 160];
  float v2 = (float)projT[base + (size_t)64 * 160];
  float v3 = (float)projT[base + (size_t)65 * 160];
  gmat[((size_t)b * NC2 + c) * NM + m] = (_Float16)fmaxf(fmaxf(v0, v1), fmaxf(v2, v3));
}

// ---------------------------------------------------------------------------
// K3: attention. WG = 128 threads (4 waves), owns 32 queries x all 1024 keys.
// LDS: scores 32x1024 f32 (128KB) + beta 32x1024 f16 (64KB) = 192KB.
// Phase A: scores -> LDS (each wave covers m-range of 256; 32 WMMAs/wave),
//          next phi B-tile prefetched before WMMAs + LDS scatter.
// Phase B: exact per-row softmax, wave32 shfl_xor reductions, beta f16 -> LDS.
// Phase C: o_preT cols 0..63 (32x64) = beta (32x1024) @ g^T, 64 WMMAs/wave, with
//          next beta A-tile (LDS) and g B-tiles (global) prefetched.
// ---------------------------------------------------------------------------
__global__ void __launch_bounds__(128)
k_attn(const _Float16* __restrict__ thetaT, const _Float16* __restrict__ phiT,
       const _Float16* __restrict__ gmat, _Float16* __restrict__ opreT) {
  extern __shared__ char smem[];
  float*    sc = (float*)smem;                                   // [32][1024] f32
  _Float16* sb = (_Float16*)(smem + (size_t)32 * 1024 * 4);      // [32][1024] f16

  int tid = threadIdx.x, wave = tid >> 5, lane = tid & 31;
  int b = blockIdx.y, n_base = blockIdx.x * 32;
  int colL = lane & 15, hi = lane >> 4, rowHi = hi * 8;
  const v8f vzero = (v8f){0.f,0.f,0.f,0.f,0.f,0.f,0.f,0.f};

  // ---- Phase A: scores = theta^T * phi -------------------------------------
  A16 a0, a1;
  {
    const _Float16* p0 = thetaT + ((size_t)b * NN + n_base + colL) * 32 + hi * 16;
    const _Float16* p1 = thetaT + ((size_t)b * NN + n_base + 16 + colL) * 32 + hi * 16;
    a0.h[0] = *(const v8h*)p0;  a0.h[1] = *(const v8h*)(p0 + 8);
    a1.h[0] = *(const v8h*)p1;  a1.h[1] = *(const v8h*)(p1 + 8);
  }
  const _Float16* phibase = phiT + ((size_t)b * NM + wave * 256 + colL) * 32 + hi * 16;
  auto loadPhi = [&](int mt) -> v16h {
    return *(const v16h*)(phibase + (size_t)(mt * 16) * 32);
  };
  v16h pb_cur = loadPhi(0);
  for (int mt = 0; mt < 16; ++mt) {
    v16h pb_nxt = loadPhi((mt + 1) & 15);
    int m0 = wave * 256 + mt * 16;
    v8f d0 = wmma_f16(a0.v, pb_cur, vzero);
    v8f d1 = wmma_f16(a1.v, pb_cur, vzero);
    #pragma unroll
    for (int v = 0; v < 8; ++v) {
      sc[(rowHi + v) * 1024 + m0 + colL]      = d0[v];
      sc[(16 + rowHi + v) * 1024 + m0 + colL] = d1[v];
    }
    pb_cur = pb_nxt;
  }
  __syncthreads();

  // ---- Phase B: softmax over keys (rows wave*8 .. wave*8+7) ----------------
  for (int i = 0; i < 8; ++i) {
    int r = wave * 8 + i;
    float vv[32];
    float mx = -3.0e38f;
    #pragma unroll
    for (int j = 0; j < 32; ++j) {
      vv[j] = sc[r * 1024 + j * 32 + lane];
      mx = fmaxf(mx, vv[j]);
    }
    for (int s = 16; s > 0; s >>= 1) mx = fmaxf(mx, __shfl_xor(mx, s, 32));
    float sum = 0.f;
    #pragma unroll
    for (int j = 0; j < 32; ++j) { vv[j] = __expf(vv[j] - mx); sum += vv[j]; }
    for (int s = 16; s > 0; s >>= 1) sum += __shfl_xor(sum, s, 32);
    float inv = 1.0f / sum;
    #pragma unroll
    for (int j = 0; j < 32; ++j)
      sb[r * 1024 + j * 32 + lane] = (_Float16)(vv[j] * inv);
  }
  __syncthreads();

  // ---- Phase C: o_preT = beta @ g^T ----------------------------------------
  int nt = wave & 1;             // n-tile (0 or 1)
  int cbase = (wave >> 1) * 32;  // two c-tiles per wave
  const _Float16* sbrow = sb + (size_t)(nt * 16 + colL) * 1024 + hi * 16;
  const _Float16* gbase = gmat + ((size_t)b * NC2 + cbase + colL) * NM + hi * 16;
  auto loadBeta = [&](int ks, A16& a) {
    a.h[0] = *(const v8h*)(sbrow + ks * 32);
    a.h[1] = *(const v8h*)(sbrow + ks * 32 + 8);
  };

  v8f acc0 = vzero, acc1 = vzero;
  A16 ba_cur, ba_nxt;
  loadBeta(0, ba_cur);
  v16h g0_cur = *(const v16h*)(gbase);
  v16h g1_cur = *(const v16h*)(gbase + (size_t)16 * NM);
  for (int ks = 0; ks < 32; ++ks) {
    int ksn = (ks + 1) & 31;
    loadBeta(ksn, ba_nxt);
    v16h g0_nxt = *(const v16h*)(gbase + ksn * 32);
    v16h g1_nxt = *(const v16h*)(gbase + (size_t)16 * NM + ksn * 32);
    acc0 = wmma_f16(ba_cur.v, g0_cur, acc0);
    acc1 = wmma_f16(ba_cur.v, g1_cur, acc1);
    ba_cur = ba_nxt; g0_cur = g0_nxt; g1_cur = g1_nxt;
  }
  #pragma unroll
  for (int v = 0; v < 8; ++v) {
    int n = n_base + nt * 16 + rowHi + v;
    opreT[((size_t)b * NN + n) * NC2 + cbase + colL]      = (_Float16)acc0[v];
    opreT[((size_t)b * NN + n) * NC2 + cbase + 16 + colL] = (_Float16)acc1[v];
  }
}

// ---------------------------------------------------------------------------
// K4: out = gamma * (o_preT @ w_o^T + b_o) + x.  K=128 -> 4 WMMA steps,
// A-tiles cached in registers across all 16 co-tiles; B-tiles rotated
// (next co-tile's 4 B-tiles prefetched while current WMMAs execute).
// ---------------------------------------------------------------------------
__global__ void __launch_bounds__(128)
k_out(const _Float16* __restrict__ opreT, const _Float16* __restrict__ wo_h,
      const float* __restrict__ b_o, const float* __restrict__ gamma,
      const float* __restrict__ x, float* __restrict__ out) {
  int tid = threadIdx.x, wave = tid >> 5, lane = tid & 31;
  int b = blockIdx.y;
  int n0 = blockIdx.x * 64 + wave * 16;
  int colL = lane & 15, hi = lane >> 4, rowHi = hi * 8;
  float gm = gamma[0];
  const v8f vzero = (v8f){0.f,0.f,0.f,0.f,0.f,0.f,0.f,0.f};

  A16 a[4];
  const _Float16* ar = opreT + ((size_t)b * NN + n0 + colL) * NC2 + hi * 16;
  #pragma unroll
  for (int ks = 0; ks < 4; ++ks) {
    a[ks].h[0] = *(const v8h*)(ar + ks * 32);
    a[ks].h[1] = *(const v8h*)(ar + ks * 32 + 8);
  }

  const _Float16* wbase = wo_h + (size_t)colL * NC2 + hi * 16;
  auto loadW = [&](int ct, int ks) -> v16h {
    return *(const v16h*)(wbase + (size_t)(ct * 16) * NC2 + ks * 32);
  };

  v16h bcur[4], bnxt[4];
  #pragma unroll
  for (int ks = 0; ks < 4; ++ks) bcur[ks] = loadW(0, ks);

  for (int ct = 0; ct < 16; ++ct) {
    int ctn = (ct + 1) & 15;
    #pragma unroll
    for (int ks = 0; ks < 4; ++ks) bnxt[ks] = loadW(ctn, ks);
    v8f acc = vzero;
    #pragma unroll
    for (int ks = 0; ks < 4; ++ks) acc = wmma_f16(a[ks].v, bcur[ks], acc);
    #pragma unroll
    for (int ks = 0; ks < 4; ++ks) bcur[ks] = bnxt[ks];

    int co = ct * 16 + colL;
    float bias = b_o[co];
    #pragma unroll
    for (int v = 0; v < 8; ++v) {
      int n = n0 + rowHi + v;
      size_t idx = ((size_t)b * NC + co) * NN + n;
      out[idx] = gm * (acc[v] + bias) + x[idx];
    }
  }
}

// ---------------------------------------------------------------------------
extern "C" void kernel_launch(void* const* d_in, const int* in_sizes, int n_in,
                              void* d_out, int out_size, void* d_ws, size_t ws_size,
                              hipStream_t stream) {
  (void)in_sizes; (void)n_in; (void)out_size; (void)ws_size;
  const float* x     = (const float*)d_in[0];
  const float* wt    = (const float*)d_in[1];
  const float* bt    = (const float*)d_in[2];
  const float* wp    = (const float*)d_in[3];
  const float* bp    = (const float*)d_in[4];
  const float* wg    = (const float*)d_in[5];
  const float* bg    = (const float*)d_in[6];
  const float* wo    = (const float*)d_in[7];
  const float* bo    = (const float*)d_in[8];
  const float* gamma = (const float*)d_in[9];
  float* out = (float*)d_out;

  char* ws = (char*)d_ws;
  _Float16* w_cat  = (_Float16*)(ws + OFF_WCAT);
  _Float16* wo_h   = (_Float16*)(ws + OFF_WOH);
  float*    b_cat  = (float*)(ws + OFF_BCAT);
  _Float16* xT     = (_Float16*)(ws + OFF_XT);
  _Float16* thetaT = (_Float16*)(ws + OFF_THETA);
  _Float16* projT  = (_Float16*)(ws + OFF_PROJ);
  _Float16* phiT   = (_Float16*)(ws + OFF_PHIT);
  _Float16* gmat   = (_Float16*)(ws + OFF_G);
  _Float16* opreT  = (_Float16*)(ws + OFF_OPRE);

  k_prep_weights<<<dim3((NO * NC + 255) / 256), dim3(256), 0, stream>>>(
      wt, wp, wg, bt, bp, bg, wo, w_cat, wo_h, b_cat);

  k_transpose_x<<<dim3(NN / 32, NC / 32, NB), dim3(32, 8), 0, stream>>>(x, xT);

  k_proj<<<dim3(NN / 64, NB), dim3(128), 0, stream>>>(xT, w_cat, b_cat, thetaT, projT);

  k_pool_phi<<<dim3(NM / 8, NB), dim3(32, 8), 0, stream>>>(projT, phiT);
  k_pool_g<<<dim3(NM / 32, NC2 / 8, NB), dim3(32, 8), 0, stream>>>(projT, gmat);

  size_t shmem = (size_t)32 * 1024 * 4 + (size_t)32 * 1024 * 2;   // 192 KB
  (void)hipFuncSetAttribute(reinterpret_cast<const void*>(k_attn),
                            hipFuncAttributeMaxDynamicSharedMemorySize, (int)shmem);
  k_attn<<<dim3(NN / 32, NB), dim3(128), shmem, stream>>>(thetaT, phiT, gmat, opreT);

  k_out<<<dim3(NN / 64, NB), dim3(128), 0, stream>>>(opreT, wo_h, bo, gamma, x, out);
}
